// LocalContext_42949673366
// MI455X (gfx1250) — matrix operand order repaired and checked by reading, not compiled, not run on hardware
//
#include <hip/hip_runtime.h>
#include <hip/hip_bf16.h>
#include <math.h>

// ---------------- problem constants ----------------
#define B_    2
#define C_    32
#define H_    128
#define W_    128
#define L_    (H_ * W_)          // 16384
#define M_TOT (B_ * L_)          // 32768 rows
#define NH_   2
#define HD_   16
#define KK_   25
#define SCALE_ 0.25f             // HD^-0.5

typedef _Float16 h16 __attribute__((ext_vector_type(16)));
typedef _Float16 h8v __attribute__((ext_vector_type(8)));
typedef float    f8  __attribute__((ext_vector_type(8)));
typedef unsigned int u32x4 __attribute__((ext_vector_type(4)));
typedef int          i32x8 __attribute__((ext_vector_type(8)));
typedef int          i32x4 __attribute__((ext_vector_type(4)));

// ---------------- workspace layout (bytes) ----------------
#define WS_WQKV   0u          //  6 blocks  * 512 halves
#define WS_WFUS   8192u       // 100 blocks * 512 halves (contiguous with WPROJ)
#define WS_WPROJ  110592u     //   8 blocks
#define WS_W1     118784u     //  16 blocks (contiguous with W2)
#define WS_W2     135168u     //  16 blocks
#define WS_BIAS   151552u     // 1250 floats
#define WS_QKVF   156672u     // 32768*96 halves
#define WS_OFLAT  6448128u    // 32768*800 halves
#define WS_YBUF   58876928u   // 32768*64 floats

#define NPK_QKV   3072        // halves in packed wqkv   (NT=6,  KS=1)
#define NPK_FUS   51200       // (NT=4,  KS=25)
#define NPK_PROJ  4096        // (NT=4,  KS=2)
#define NPK_W1    8192        // (NT=8,  KS=2)
#define NPK_W2    8192        // (NT=4,  KS=4)
#define NPK_BIAS  1250
#define NPK_TOTAL (NPK_QKV + NPK_FUS + NPK_PROJ + NPK_W1 + NPK_W2 + NPK_BIAS)

#define WMMA_F16(A, Bm, Cm) \
  __builtin_amdgcn_wmma_f32_16x16x32_f16(false, (A), false, (Bm), (short)0, (Cm), false, false)

__device__ __forceinline__ h16 frag2(const _Float16* p0, const _Float16* p1) {
  union { h16 v; h8v h[2]; } u;
  u.h[0] = *(const h8v*)p0;
  u.h[1] = *(const h8v*)p1;
  return u.v;
}

// ----------------------------------------------------------------------
// TDM: DMA `nhalves` f16 elements from global to LDS (one 1-row 2D tile).
// D# per CDNA5 ISA ch.8: group0 = {count/type/addresses}, group1 = dims.
// Issued by one full wave; completion via s_wait_tensorcnt, then WG barrier.
// ----------------------------------------------------------------------
__device__ __forceinline__ void tdm_load_to_lds(unsigned lds_off, const void* gptr,
                                                unsigned nhalves) {
  unsigned long long ga = (unsigned long long)(uintptr_t)gptr;
  u32x4 g0;
  g0[0] = 1u;                                              // count=1, user mode
  g0[1] = lds_off;                                         // lds_addr (bytes)
  g0[2] = (unsigned)(ga & 0xFFFFFFFFu);                    // global_addr[31:0]
  g0[3] = (unsigned)((ga >> 32) & 0x1FFFFFFu) | (2u << 30); // addr[56:32] | type=2
  i32x8 g1;
  g1[0] = 1 << 16;                                         // data_size=1 (2 bytes)
  g1[1] = (int)((nhalves & 0xFFFFu) << 16);                // tensor_dim0[15:0]
  g1[2] = (int)((nhalves >> 16) | (1u << 16));             // dim0[31:16] | tensor_dim1=1
  g1[3] = (int)((nhalves & 0xFFFFu) << 16);                // tile_dim0
  g1[4] = 1;                                               // tile_dim1=1, tile_dim2=0
  g1[5] = (int)nhalves;                                    // tensor_dim0_stride[31:0]
  g1[6] = 0;
  g1[7] = 0;
  i32x4 z4 = {0, 0, 0, 0};
#if __clang_major__ >= 23
  i32x8 z8 = {0, 0, 0, 0, 0, 0, 0, 0};
  __builtin_amdgcn_tensor_load_to_lds(g0, g1, z4, z4, z8, 0);
#else
  __builtin_amdgcn_tensor_load_to_lds(g0, g1, z4, z4, 0);
#endif
}

// ======================================================================
// Kernel 0: repack weights into WMMA B-fragment order (f16) + bias table.
// Packed layout: [((ks*NT + nt)*32 + lane)*16 + e]
//   B element (k, n):  k = ks*32 + ((lane&16)?8:0) + (e&7) + ((e&8)?16:0) ;  n = nt*16 + (lane&15)
// ======================================================================
__global__ void k_prep(const float* __restrict__ qkv_w, const float* __restrict__ fusion_w,
                       const float* __restrict__ proj_w, const float* __restrict__ mlp_w1,
                       const float* __restrict__ mlp_w2, const float* __restrict__ rel_table,
                       char* __restrict__ ws) {
  int gid = blockIdx.x * blockDim.x + threadIdx.x;
  if (gid >= NPK_TOTAL) return;

  _Float16* wqkv  = (_Float16*)(ws + WS_WQKV);
  _Float16* wfus  = (_Float16*)(ws + WS_WFUS);
  _Float16* wproj = (_Float16*)(ws + WS_WPROJ);
  _Float16* w1p   = (_Float16*)(ws + WS_W1);
  _Float16* w2p   = (_Float16*)(ws + WS_W2);
  float*    biasT = (float*)(ws + WS_BIAS);

  int idx = gid;
  if (idx < NPK_QKV) {
    int blk = idx >> 9, w = idx & 511, lane = w >> 4, e = w & 15;
    int nt = blk % 6;                       // KS = 1
    int k = ((lane & 16) ? 8 : 0) + (e & 7) + ((e & 8) ? 16 : 0);     // 0..31  (channel c)
    int n = nt * 16 + (lane & 15);          // 0..95  (output o)
    wqkv[idx] = (_Float16)qkv_w[n * 32 + k];
    return;
  }
  idx -= NPK_QKV;
  if (idx < NPK_FUS) {
    int blk = idx >> 9, w = idx & 511, lane = w >> 4, e = w & 15;
    int ks = blk / 4, nt = blk % 4;
    int k = ks * 32 + ((lane & 16) ? 8 : 0) + (e & 7) + ((e & 8) ? 16 : 0);   // 0..799 = p*32 + c
    int n = nt * 16 + (lane & 15);                  // 0..63
    int p = k >> 5, c = k & 31, i = p / 5, j = p % 5;
    wfus[idx] = (_Float16)fusion_w[((n * 32 + c) * 5 + i) * 5 + j];
    return;
  }
  idx -= NPK_FUS;
  if (idx < NPK_PROJ) {
    int blk = idx >> 9, w = idx & 511, lane = w >> 4, e = w & 15;
    int ks = blk / 4, nt = blk % 4;
    int k = ks * 32 + ((lane & 16) ? 8 : 0) + (e & 7) + ((e & 8) ? 16 : 0);   // 0..63
    int n = nt * 16 + (lane & 15);
    wproj[idx] = (_Float16)proj_w[n * 64 + k];
    return;
  }
  idx -= NPK_PROJ;
  if (idx < NPK_W1) {
    int blk = idx >> 9, w = idx & 511, lane = w >> 4, e = w & 15;
    int ks = blk / 8, nt = blk % 8;
    int k = ks * 32 + ((lane & 16) ? 8 : 0) + (e & 7) + ((e & 8) ? 16 : 0);   // 0..63
    int n = nt * 16 + (lane & 15);                  // 0..127
    w1p[idx] = (_Float16)mlp_w1[n * 64 + k];
    return;
  }
  idx -= NPK_W1;
  if (idx < NPK_W2) {
    int blk = idx >> 9, w = idx & 511, lane = w >> 4, e = w & 15;
    int ks = blk / 4, nt = blk % 4;
    int k = ks * 32 + ((lane & 16) ? 8 : 0) + (e & 7) + ((e & 8) ? 16 : 0);   // 0..127
    int n = nt * 16 + (lane & 15);                  // 0..63
    w2p[idx] = (_Float16)mlp_w2[n * 128 + k];
    return;
  }
  idx -= NPK_W2;
  // relative-position bias table: biasT[(h*25 + q)*25 + kk]
  {
    int h = idx / 625, rem = idx % 625, q = rem / 25, kk = rem % 25;
    int qi = q / 5, qj = q % 5, ki = kk / 5, kj = kk % 5;
    int relidx = (qi - ki + 4) * 9 + (qj - kj + 4);
    biasT[idx] = rel_table[relidx * NH_ + h];
  }
}

// ======================================================================
// Kernel 1: LN1 + QKV GEMM  (M=32768, K=32, N=96).  128 rows / WG, 8 waves.
// ======================================================================
__global__ void __launch_bounds__(256)
k_ln_qkv(const float* __restrict__ x, const float* __restrict__ qkv_b,
         const float* __restrict__ n1w, const float* __restrict__ n1b,
         const _Float16* __restrict__ wqkv, _Float16* __restrict__ qkvf) {
  __shared__ _Float16 sA[128 * 32];
  const int t = threadIdx.x;
  const int rowBase = blockIdx.x * 128;

  if (t < 128) {
    int m = rowBase + t;
    int b = m >> 14, l = m & (L_ - 1);
    const float* xp = x + (size_t)b * C_ * L_ + l;
    float vals[32];
    float mean = 0.f;
#pragma unroll
    for (int c = 0; c < 32; ++c) { vals[c] = xp[(size_t)c * L_]; mean += vals[c]; }
    mean *= (1.f / 32.f);
    float var = 0.f;
#pragma unroll
    for (int c = 0; c < 32; ++c) { float d = vals[c] - mean; var += d * d; }
    var *= (1.f / 32.f);
    float inv = rsqrtf(var + 1e-5f);
#pragma unroll
    for (int c = 0; c < 32; ++c)
      sA[t * 32 + c] = (_Float16)((vals[c] - mean) * inv * n1w[c] + n1b[c]);
  }
  __syncthreads();

  const int lane = t & 31, wv = t >> 5;
  const int l15 = lane & 15, hi = (lane >> 4) & 1;
  const int koff = hi * 8;

  const _Float16* a0 = &sA[(wv * 16 + l15) * 32 + koff];
  h16 a = frag2(a0, a0 + 16);

  f8 acc[6];
#pragma unroll
  for (int nt = 0; nt < 6; ++nt) {
    const _Float16* bp = wqkv + ((size_t)(nt * 32 + lane)) * 16;
    h16 bf = frag2(bp, bp + 8);
    f8 cz = {};
    acc[nt] = WMMA_F16(a, bf, cz);
  }

  const int mrow0 = rowBase + wv * 16 + hi * 8;
#pragma unroll
  for (int nt = 0; nt < 6; ++nt) {
    int n = nt * 16 + l15;
    float bb = qkv_b[n];
#pragma unroll
    for (int r = 0; r < 8; ++r) {
      int m = mrow0 + r;
      qkvf[(size_t)m * 96 + n] = (_Float16)(acc[nt][r] + bb);
    }
  }
}

// ======================================================================
// Kernel 2: checkerboard local-window attention.  1 thread = (pixel, head).
// Writes o windows f16: [((b*L + l)*25 + p)*32 + h*16 + d]
// ======================================================================
__global__ void __launch_bounds__(256)
k_attn(const _Float16* __restrict__ qkvf, const float* __restrict__ biasT,
       _Float16* __restrict__ oflat) {
  int gid = blockIdx.x * 256 + threadIdx.x;
  int h = gid & 1;
  int pix = gid >> 1;
  int b = pix >> 14, l = pix & (L_ - 1);
  int y0 = l >> 7, x0 = l & (W_ - 1);
  int par0 = (y0 + x0) & 1;

  const size_t rowstride = 96;
  const _Float16* base = qkvf + (size_t)(b * L_) * rowstride;

  for (int qi = 0; qi < KK_; ++qi) {
    int qdi = qi / 5 - 2, qdj = qi % 5 - 2;
    int qy = y0 + qdi, qx = x0 + qdj;
    bool qin = (unsigned)qy < H_ && (unsigned)qx < W_;
    bool cwq = qin && (((par0 + qdi + qdj) & 1) == 1);

    float qr[16];
    if (qin) {
      const _Float16* qp = base + (size_t)(qy * W_ + qx) * rowstride + h * HD_;
      h8v q0 = *(const h8v*)qp, q1 = *(const h8v*)(qp + 8);
#pragma unroll
      for (int d = 0; d < 8; ++d) { qr[d] = (float)q0[d]; qr[8 + d] = (float)q1[d]; }
    } else {
#pragma unroll
      for (int d = 0; d < 16; ++d) qr[d] = 0.f;
    }

    float s[KK_];
    float smax = -1e30f;
    for (int ki = 0; ki < KK_; ++ki) {
      int kdi = ki / 5 - 2, kdj = ki % 5 - 2;
      int ky = y0 + kdi, kx = x0 + kdj;
      bool kin = (unsigned)ky < H_ && (unsigned)kx < W_;
      bool cwk = kin && (((par0 + kdi + kdj) & 1) == 1);
      float dot = 0.f;
      if (kin) {
        const _Float16* kp = base + (size_t)(ky * W_ + kx) * rowstride + 32 + h * HD_;
        h8v k0 = *(const h8v*)kp, k1 = *(const h8v*)(kp + 8);
#pragma unroll
        for (int d = 0; d < 8; ++d) dot += qr[d] * (float)k0[d] + qr[8 + d] * (float)k1[d];
      }
      float msk = (cwq && cwk) ? 0.f : -100.f;
      float sc = dot * SCALE_ + biasT[(h * 25 + qi) * 25 + ki] + msk;
      s[ki] = sc;
      smax = fmaxf(smax, sc);
    }
    float sum = 0.f;
#pragma unroll
    for (int ki = 0; ki < KK_; ++ki) { s[ki] = __expf(s[ki] - smax); sum += s[ki]; }
    float inv = 1.f / sum;

    float acc[16];
#pragma unroll
    for (int d = 0; d < 16; ++d) acc[d] = 0.f;
    for (int ki = 0; ki < KK_; ++ki) {
      int kdi = ki / 5 - 2, kdj = ki % 5 - 2;
      int ky = y0 + kdi, kx = x0 + kdj;
      if ((unsigned)ky < H_ && (unsigned)kx < W_) {
        const _Float16* vp = base + (size_t)(ky * W_ + kx) * rowstride + 64 + h * HD_;
        h8v v0 = *(const h8v*)vp, v1 = *(const h8v*)(vp + 8);
        float w = s[ki] * inv;
#pragma unroll
        for (int d = 0; d < 8; ++d) { acc[d] += w * (float)v0[d]; acc[8 + d] += w * (float)v1[d]; }
      }
    }

    _Float16* op = oflat + (((size_t)(b * L_ + l) * 25 + qi) * 32 + h * HD_);
    h8v o0, o1;
#pragma unroll
    for (int d = 0; d < 8; ++d) { o0[d] = (_Float16)acc[d]; o1[d] = (_Float16)acc[8 + d]; }
    *(h8v*)op = o0;
    *(h8v*)(op + 8) = o1;
  }
}

// ======================================================================
// Kernel 3: fusion GEMM (K=800, N=64) fused with proj GEMM (K=64, N=64).
// TDM stages packed Wfus+Wproj (55296 halves, 108 KB) into LDS once per WG;
// all B fragments then come from ds_load_b128.
// ======================================================================
__global__ void __launch_bounds__(256)
k_fusion_proj(const _Float16* __restrict__ oflat, const _Float16* __restrict__ wfus,
              const float* __restrict__ fus_b, const float* __restrict__ proj_b,
              float* __restrict__ ybuf) {
  __shared__ _Float16 sW[NPK_FUS + NPK_PROJ];   // 55296 halves = 108 KB
  __shared__ _Float16 sY[8 * 16 * 64];
  const int t = threadIdx.x;
  const int lane = t & 31, wv = t >> 5;
  const int l15 = lane & 15, hi = (lane >> 4) & 1;
  const int koff = hi * 8;
  const int mrow = blockIdx.x * 128 + wv * 16;

  if (wv == 0) {    // one wave issues the tensor DMA for the whole WG
    tdm_load_to_lds((unsigned)(uintptr_t)&sW[0], wfus, NPK_FUS + NPK_PROJ);
    __builtin_amdgcn_s_wait_tensorcnt(0);
  }
  __syncthreads();

  const _Float16* arow = oflat + (size_t)(mrow + l15) * 800 + koff;

  f8 acc[4] = {};
  for (int ks = 0; ks < 25; ++ks) {
    __builtin_prefetch(arow + (ks + 1) * 32, 0, 0);
    h16 a = frag2(arow + ks * 32, arow + ks * 32 + 16);
#pragma unroll
    for (int nt = 0; nt < 4; ++nt) {
      const _Float16* bp = &sW[((ks * 4 + nt) * 32 + lane) * 16];
      h16 bf = frag2(bp, bp + 8);
      acc[nt] = WMMA_F16(a, bf, acc[nt]);
    }
  }

  // y1 (+fusion bias) -> LDS in f16 for the proj GEMM
  _Float16* yt = &sY[wv * 16 * 64];
#pragma unroll
  for (int nt = 0; nt < 4; ++nt) {
    int n = nt * 16 + l15;
    float bb = fus_b[n];
#pragma unroll
    for (int r = 0; r < 8; ++r)
      yt[(hi * 8 + r) * 64 + n] = (_Float16)(acc[nt][r] + bb);
  }
  __syncthreads();

  f8 acc2[4] = {};
#pragma unroll
  for (int ks = 0; ks < 2; ++ks) {
    const _Float16* a0 = &yt[l15 * 64 + ks * 32 + koff];
    h16 a = frag2(a0, a0 + 16);
#pragma unroll
    for (int nt = 0; nt < 4; ++nt) {
      const _Float16* bp = &sW[NPK_FUS + ((ks * 4 + nt) * 32 + lane) * 16];
      h16 bf = frag2(bp, bp + 8);
      acc2[nt] = WMMA_F16(a, bf, acc2[nt]);
    }
  }

#pragma unroll
  for (int nt = 0; nt < 4; ++nt) {
    int n = nt * 16 + l15;
    float bb = proj_b[n];
#pragma unroll
    for (int r = 0; r < 8; ++r) {
      int m = mrow + hi * 8 + r;
      ybuf[(size_t)m * 64 + n] = acc2[nt][r] + bb;
    }
  }
}

// ======================================================================
// Kernel 4: LN2 -> MLP (64->128 GELU ->64) -> residual -> transposed store.
// TDM stages packed W1+W2 (16384 halves, 32 KB) into LDS once per WG.
// ======================================================================
__global__ void __launch_bounds__(256)
k_ln2_mlp(const float* __restrict__ ybuf, const float* __restrict__ n2w,
          const float* __restrict__ n2b, const _Float16* __restrict__ w1p,
          const float* __restrict__ b1, const float* __restrict__ b2,
          float* __restrict__ out) {
  __shared__ _Float16 sW[NPK_W1 + NPK_W2];   // 16384 halves = 32 KB
  __shared__ _Float16 sA1[128 * 64];
  __shared__ _Float16 sA2[8 * 16 * 128];
  const int t = threadIdx.x;
  const int rowBase = blockIdx.x * 128;

  if ((t >> 5) == 0) {
    tdm_load_to_lds((unsigned)(uintptr_t)&sW[0], w1p, NPK_W1 + NPK_W2);
    __builtin_amdgcn_s_wait_tensorcnt(0);
  }

  if (t < 128) {
    int m = rowBase + t;
    const float* yp = ybuf + (size_t)m * 64;
    float s = 0.f, s2 = 0.f;
#pragma unroll
    for (int c = 0; c < 64; ++c) { float v = yp[c]; s += v; s2 += v * v; }
    float mean = s * (1.f / 64.f);
    float var = s2 * (1.f / 64.f) - mean * mean;
    float inv = rsqrtf(var + 1e-5f);
#pragma unroll
    for (int c = 0; c < 64; ++c)
      sA1[t * 64 + c] = (_Float16)((yp[c] - mean) * inv * n2w[c] + n2b[c]);
  }
  __syncthreads();

  const int lane = t & 31, wv = t >> 5;
  const int l15 = lane & 15, hi = (lane >> 4) & 1;
  const int koff = hi * 8;

  // GEMM3: 64 -> 128
  f8 acc[8] = {};
#pragma unroll
  for (int ks = 0; ks < 2; ++ks) {
    const _Float16* a0 = &sA1[(wv * 16 + l15) * 64 + ks * 32 + koff];
    h16 a = frag2(a0, a0 + 16);
#pragma unroll
    for (int nt = 0; nt < 8; ++nt) {
      const _Float16* bp = &sW[((ks * 8 + nt) * 32 + lane) * 16];
      h16 bf = frag2(bp, bp + 8);
      acc[nt] = WMMA_F16(a, bf, acc[nt]);
    }
  }

  // exact GELU -> LDS f16
  _Float16* a2 = &sA2[wv * 16 * 128];
#pragma unroll
  for (int nt = 0; nt < 8; ++nt) {
    int n = nt * 16 + l15;
    float bb = b1[n];
#pragma unroll
    for (int r = 0; r < 8; ++r) {
      float v = acc[nt][r] + bb;
      float g = 0.5f * v * (1.f + erff(v * 0.70710678f));
      a2[(hi * 8 + r) * 128 + n] = (_Float16)g;
    }
  }
  __syncthreads();

  // GEMM4: 128 -> 64
  f8 acc2[4] = {};
#pragma unroll
  for (int ks = 0; ks < 4; ++ks) {
    const _Float16* a0 = &a2[l15 * 128 + ks * 32 + koff];
    h16 a = frag2(a0, a0 + 16);
#pragma unroll
    for (int nt = 0; nt < 4; ++nt) {
      const _Float16* bp = &sW[NPK_W1 + ((ks * 4 + nt) * 32 + lane) * 16];
      h16 bf = frag2(bp, bp + 8);
      acc2[nt] = WMMA_F16(a, bf, acc2[nt]);
    }
  }

#pragma unroll
  for (int nt = 0; nt < 4; ++nt) {
    int n = nt * 16 + l15;
    float bb = b2[n];
#pragma unroll
    for (int r = 0; r < 8; ++r) {
      int m = rowBase + wv * 16 + hi * 8 + r;
      int b = m >> 14, l = m & (L_ - 1);
      float v = acc2[nt][r] + bb + ybuf[(size_t)m * 64 + n];
      out[(size_t)b * 64 * L_ + (size_t)n * L_ + l] = v;
    }
  }
}

// ======================================================================
extern "C" void kernel_launch(void* const* d_in, const int* in_sizes, int n_in,
                              void* d_out, int out_size, void* d_ws, size_t ws_size,
                              hipStream_t stream) {
  const float* x        = (const float*)d_in[0];
  const float* qkv_w    = (const float*)d_in[1];
  const float* qkv_b    = (const float*)d_in[2];
  const float* rel_tab  = (const float*)d_in[3];
  const float* n1w      = (const float*)d_in[4];
  const float* n1b      = (const float*)d_in[5];
  const float* n2w      = (const float*)d_in[6];
  const float* n2b      = (const float*)d_in[7];
  const float* proj_w   = (const float*)d_in[8];
  const float* proj_b   = (const float*)d_in[9];
  const float* fusion_w = (const float*)d_in[10];
  const float* fusion_b = (const float*)d_in[11];
  const float* mlp_w1   = (const float*)d_in[12];
  const float* mlp_b1   = (const float*)d_in[13];
  const float* mlp_w2   = (const float*)d_in[14];
  const float* mlp_b2   = (const float*)d_in[15];

  char* ws = (char*)d_ws;
  _Float16* wqkv  = (_Float16*)(ws + WS_WQKV);
  _Float16* wfus  = (_Float16*)(ws + WS_WFUS);
  _Float16* w1p   = (_Float16*)(ws + WS_W1);
  float*    biasT = (float*)(ws + WS_BIAS);
  _Float16* qkvf  = (_Float16*)(ws + WS_QKVF);
  _Float16* oflat = (_Float16*)(ws + WS_OFLAT);
  float*    ybuf  = (float*)(ws + WS_YBUF);

  float* out = (float*)d_out;

  k_prep<<<(NPK_TOTAL + 255) / 256, 256, 0, stream>>>(qkv_w, fusion_w, proj_w,
                                                      mlp_w1, mlp_w2, rel_tab, ws);
  k_ln_qkv<<<M_TOT / 128, 256, 0, stream>>>(x, qkv_b, n1w, n1b, wqkv, qkvf);
  k_attn<<<(M_TOT * NH_) / 256, 256, 0, stream>>>(qkvf, biasT, oflat);
  k_fusion_proj<<<M_TOT / 128, 256, 0, stream>>>(oflat, wfus, fusion_b, proj_b, ybuf);
  k_ln2_mlp<<<M_TOT / 128, 256, 0, stream>>>(ybuf, n2w, n2b, w1p, mlp_b1, mlp_b2, out);
}
